// DeformableLocalCorrAggr_82025285419164
// MI455X (gfx1250) — matrix operand order repaired and checked by reading, not compiled, not run on hardware
//
#include <hip/hip_runtime.h>

typedef __attribute__((ext_vector_type(16))) _Float16 v16h;
typedef __attribute__((ext_vector_type(8)))  float    v8f;

static constexpr int BB = 4;
static constexpr int CC = 64;
static constexpr int HH = 128;
static constexpr int WW = 128;
static constexpr int NPIX = BB * HH * WW;      // 65536
static constexpr int CATC = 4 * CC;            // 256

// ---------------------------------------------------------------------------
// WMMA fragment loaders (CDNA5 16x16x32 f16, wave32, contiguous k = 16h + s in both A and B)
// ---------------------------------------------------------------------------

// A matrix 16x32 (MxK), row-major source with given row stride (halves).
// lanes 0-15: M=lane, K=0..15; lanes 16-31: M=lane-16, K=16..31.
__device__ __forceinline__ v16h load_A(const _Float16* base, int stride, int lane) {
  int m   = lane & 15;
  int khi = (lane & 16) ? 16 : 0;
  const _Float16* p = base + m * stride + khi;
  v16h a;
#pragma unroll
  for (int j = 0; j < 8; ++j) {
    int k0 = 2 * j;
    a[2 * j]     = p[k0];
    a[2 * j + 1] = p[k0 + 1];
  }
  return a;
}

// B matrix 32x16 (KxN) from packed slab laid out [cout][32 k] (halves):
// lanes 0-15: N=lane, K=0..15; lanes 16-31: N=lane-16, K=16..31.
// Each lane's fragment is one contiguous 32B vector.
__device__ __forceinline__ v16h load_B(const _Float16* slab, int lane) {
  int n  = lane & 15;
  int kb = (lane & 16) ? 16 : 0;
  return *(const v16h*)(slab + n * 32 + kb);
}

// ---------------------------------------------------------------------------
// Kernel 1: NCHW f32 -> NHWC f16 packing for Q, K(=sup) and concat block 0
// ---------------------------------------------------------------------------
__global__ void pack_inputs(const float* __restrict__ sup, const float* __restrict__ key,
                            _Float16* __restrict__ Kh, _Float16* __restrict__ Qh,
                            _Float16* __restrict__ cat) {
  int idx = blockIdx.x * blockDim.x + threadIdx.x;
  if (idx >= BB * CC * HH * WW) return;
  int x = idx % WW;
  int y = (idx / WW) % HH;
  int c = (idx / (WW * HH)) % CC;
  int b = idx / (WW * HH * CC);
  float sv = sup[idx];
  float kv = key[idx];
  long pix = (long)(b * HH + y) * WW + x;
  Kh[pix * CC + c]   = (_Float16)sv;
  Qh[pix * CC + c]   = (_Float16)kv;
  cat[pix * CATC + c] = (_Float16)sv;
}

// ---------------------------------------------------------------------------
// Kernel 2: repack OIHW f32 weights -> f16 [tap][I/32][O][32]
// ---------------------------------------------------------------------------
__global__ void repack_weights(const float* __restrict__ w, _Float16* __restrict__ out,
                               int O, int I, int taps) {
  int idx = blockIdx.x * blockDim.x + threadIdx.x;
  if (idx >= O * I * taps) return;
  int t = idx % taps;
  int i = (idx / taps) % I;
  int o = idx / (taps * I);
  out[((t * (I >> 5) + (i >> 5)) * O + o) * 32 + (i & 31)] = (_Float16)w[idx];
}

// ---------------------------------------------------------------------------
// Kernel 3: 1x1 conv (per-pixel 64x64 GEMM) via WMMA.
// Block = 128 threads (4 waves); tile = 16 pixels x 64 cout.
// ---------------------------------------------------------------------------
__global__ void conv1x1_wmma(const _Float16* __restrict__ X, const _Float16* __restrict__ Wp,
                             const float* __restrict__ bias, _Float16* __restrict__ Y) {
  __shared__ __align__(16) _Float16 ldsA[16 * 64];
  int tid = threadIdx.x, lane = tid & 31, wv = tid >> 5;
  long pixbase = (long)blockIdx.x * 16;

  // stage 16 pixels x 64 ch (contiguous in NHWC) into LDS
  const uint* src = (const uint*)(X + pixbase * CC);
  uint* dst = (uint*)ldsA;
#pragma unroll
  for (int j = tid; j < 512; j += 128) dst[j] = src[j];
  __syncthreads();

  v8f acc = {};
#pragma unroll
  for (int kc = 0; kc < 2; ++kc) {
    v16h a = load_A(ldsA + kc * 32, CC, lane);
    v16h b = load_B(Wp + (kc * 64 + wv * 16) * 32, lane);
    acc = __builtin_amdgcn_wmma_f32_16x16x32_f16(false, a, false, b, (short)0, acc, false, false);
  }
  int n = lane & 15, hi = (lane & 16) ? 8 : 0;
  int cout = wv * 16 + n;
  float bs = bias[cout];
#pragma unroll
  for (int r = 0; r < 8; ++r)
    Y[(pixbase + r + hi) * CC + cout] = (_Float16)(acc[r] + bs);
}

// ---------------------------------------------------------------------------
// Kernel 4: neighborhood attention (k in {3,5,7}), one thread per pixel.
// fp32 VALU math on f16 data; writes f16 into concat channel block.
// ---------------------------------------------------------------------------
template <int K>
__global__ void natten_kernel(const _Float16* __restrict__ Qh, const _Float16* __restrict__ Kh,
                              const _Float16* __restrict__ Vh, const float* __restrict__ rpb,
                              _Float16* __restrict__ cat, int chanBase) {
  constexpr int N = K / 2;
  constexpr int R = 2 * K - 1;
  int idx = blockIdx.x * blockDim.x + threadIdx.x;
  if (idx >= NPIX) return;
  int x = idx % WW;
  int y = (idx / WW) % HH;
  int b = idx / (WW * HH);

  int sy = min(max(y - N, 0), HH - K);
  int sx = min(max(x - N, 0), WW - K);
  int pby = N + max(N - y, 0) + ((y + N >= HH) ? (HH - y - 1 - N) : 0);
  int pbx = N + max(N - x, 0) + ((x + N >= WW) ? (WW - x - 1 - N) : 0);

  float logits[K * K];
#pragma unroll
  for (int pq = 0; pq < K * K; ++pq)
    logits[pq] = rpb[(pby + pq / K) * R + (pbx + pq % K)];

  const _Float16* qp = Qh + (long)idx * CC;
  for (int cb = 0; cb < CC; cb += 8) {
    float qv[8];
#pragma unroll
    for (int t = 0; t < 8; ++t) qv[t] = (float)qp[cb + t];
#pragma unroll
    for (int pq = 0; pq < K * K; ++pq) {
      int gy = sy + pq / K, gx = sx + pq % K;
      const _Float16* kp = Kh + ((long)(b * HH + gy) * WW + gx) * CC + cb;
      float d = 0.f;
#pragma unroll
      for (int t = 0; t < 8; ++t) d += qv[t] * (float)kp[t];
      logits[pq] += d;
    }
  }

  float mx = logits[0];
#pragma unroll
  for (int pq = 1; pq < K * K; ++pq) mx = fmaxf(mx, logits[pq]);
  float sum = 0.f;
#pragma unroll
  for (int pq = 0; pq < K * K; ++pq) {
    logits[pq] = __expf(logits[pq] - mx);
    sum += logits[pq];
  }
  float inv = 1.f / sum;

  _Float16* op = cat + (long)idx * CATC + chanBase;
  for (int cb = 0; cb < CC; cb += 8) {
    float acc[8] = {0.f, 0.f, 0.f, 0.f, 0.f, 0.f, 0.f, 0.f};
#pragma unroll
    for (int pq = 0; pq < K * K; ++pq) {
      int gy = sy + pq / K, gx = sx + pq % K;
      const _Float16* vp = Vh + ((long)(b * HH + gy) * WW + gx) * CC + cb;
      float w = logits[pq];
#pragma unroll
      for (int t = 0; t < 8; ++t) acc[t] += w * (float)vp[t];
    }
#pragma unroll
    for (int t = 0; t < 8; ++t) op[cb + t] = (_Float16)(acc[t] * inv);
  }
}

// ---------------------------------------------------------------------------
// Kernel 5: 3x3 SAME conv via implicit GEMM + WMMA.
// Block = 128 threads (4 waves); tile = 16 pixels (one row run) x 64 cout.
// CIN=256 (wf1, relu -> f16 NHWC mid) or CIN=64 (wf2, relu -> f32 NCHW out).
// ---------------------------------------------------------------------------
template <int CIN, bool FINAL>
__global__ void conv3x3_wmma(const _Float16* __restrict__ X, const _Float16* __restrict__ Wp,
                             const float* __restrict__ bias, _Float16* __restrict__ Ymid,
                             float* __restrict__ Yout) {
  constexpr int KC = CIN / 32;
  constexpr int VPS = CIN / 8;  // uint4 per halo slot
  __shared__ __align__(16) _Float16 lds[3 * 18 * CIN];

  int tid = threadIdx.x, lane = tid & 31, wv = tid >> 5;
  int t = blockIdx.x;
  int x0 = (t & 7) * 16;
  int y = (t >> 3) & 127;
  int b = t >> 10;

  // stage 3 rows x 18 cols halo with zero padding
  for (int j = tid; j < 54 * VPS; j += 128) {
    int slot = j / VPS, off = j % VPS;
    int dy = slot / 18, col = slot % 18;
    int gy = y + dy - 1, gx = x0 + col - 1;
    uint4 v;
    v.x = v.y = v.z = v.w = 0u;
    if (gy >= 0 && gy < HH && gx >= 0 && gx < WW)
      v = ((const uint4*)(X + ((long)(b * HH + gy) * WW + gx) * CIN))[off];
    ((uint4*)(lds + slot * CIN))[off] = v;
  }
  __syncthreads();

  v8f acc = {};
#pragma unroll
  for (int tap = 0; tap < 9; ++tap) {
    int ky = tap / 3, kx = tap % 3;
#pragma unroll
    for (int kc = 0; kc < KC; ++kc) {
      v16h a  = load_A(lds + (ky * 18 + kx) * CIN + kc * 32, CIN, lane);
      v16h bm = load_B(Wp + ((tap * KC + kc) * 64 + wv * 16) * 32, lane);
      acc = __builtin_amdgcn_wmma_f32_16x16x32_f16(false, a, false, bm, (short)0, acc, false, false);
    }
  }

  int n = lane & 15, hi = (lane & 16) ? 8 : 0;
  int cout = wv * 16 + n;
  float bs = bias[cout];
#pragma unroll
  for (int r = 0; r < 8; ++r) {
    float v = fmaxf(acc[r] + bs, 0.f);
    int px = x0 + r + hi;
    if constexpr (FINAL) {
      Yout[((long)(b * CC + cout) * HH + y) * WW + px] = v;
    } else {
      Ymid[((long)(b * HH + y) * WW + px) * CC + cout] = (_Float16)v;
    }
  }
}

// ---------------------------------------------------------------------------
// Launch
// ---------------------------------------------------------------------------
extern "C" void kernel_launch(void* const* d_in, const int* in_sizes, int n_in,
                              void* d_out, int out_size, void* d_ws, size_t ws_size,
                              hipStream_t stream) {
  (void)in_sizes; (void)n_in; (void)out_size; (void)ws_size;

  const float* sup = (const float*)d_in[0];
  const float* key = (const float*)d_in[1];
  const float* rpb[3] = {(const float*)d_in[2], (const float*)d_in[3], (const float*)d_in[4]};
  const float* wv[3]  = {(const float*)d_in[5], (const float*)d_in[7], (const float*)d_in[9]};
  const float* bv[3]  = {(const float*)d_in[6], (const float*)d_in[8], (const float*)d_in[10]};
  const float* wf1 = (const float*)d_in[11];
  const float* bf1 = (const float*)d_in[12];
  const float* wf2 = (const float*)d_in[13];
  const float* bf2 = (const float*)d_in[14];

  char* ws = (char*)d_ws;
  size_t off = 0;
  auto carve = [&](size_t bytes) -> void* {
    void* p = ws + off;
    off += (bytes + 255) & ~(size_t)255;
    return p;
  };
  _Float16* Qh  = (_Float16*)carve((size_t)NPIX * CC * 2);
  _Float16* Kh  = (_Float16*)carve((size_t)NPIX * CC * 2);
  _Float16* Vh0 = (_Float16*)carve((size_t)NPIX * CC * 2);
  _Float16* Vh1 = (_Float16*)carve((size_t)NPIX * CC * 2);
  _Float16* Vh2 = (_Float16*)carve((size_t)NPIX * CC * 2);
  _Float16* Vh[3] = {Vh0, Vh1, Vh2};
  _Float16* cat = (_Float16*)carve((size_t)NPIX * CATC * 2);
  _Float16* mid = (_Float16*)carve((size_t)NPIX * CC * 2);
  _Float16* wvp[3];
  for (int i = 0; i < 3; ++i) wvp[i] = (_Float16*)carve((size_t)64 * 64 * 2);
  _Float16* wf1p = (_Float16*)carve((size_t)9 * 256 * 64 * 2);
  _Float16* wf2p = (_Float16*)carve((size_t)9 * 64 * 64 * 2);

  // 1) pack inputs
  pack_inputs<<<(BB * CC * HH * WW + 255) / 256, 256, 0, stream>>>(sup, key, Kh, Qh, cat);

  // 2) repack weights
  for (int i = 0; i < 3; ++i)
    repack_weights<<<(64 * 64 + 255) / 256, 256, 0, stream>>>(wv[i], wvp[i], 64, 64, 1);
  repack_weights<<<(64 * 256 * 9 + 255) / 256, 256, 0, stream>>>(wf1, wf1p, 64, 256, 9);
  repack_weights<<<(64 * 64 * 9 + 255) / 256, 256, 0, stream>>>(wf2, wf2p, 64, 64, 9);

  // 3) V projections (1x1 conv, WMMA)
  for (int i = 0; i < 3; ++i)
    conv1x1_wmma<<<NPIX / 16, 128, 0, stream>>>(Kh, wvp[i], bv[i], Vh[i]);

  // 4) neighborhood attention per level -> concat channel blocks
  natten_kernel<3><<<NPIX / 256, 256, 0, stream>>>(Qh, Kh, Vh[0], rpb[0], cat, 64);
  natten_kernel<5><<<NPIX / 256, 256, 0, stream>>>(Qh, Kh, Vh[1], rpb[1], cat, 128);
  natten_kernel<7><<<NPIX / 256, 256, 0, stream>>>(Qh, Kh, Vh[2], rpb[2], cat, 192);

  // 5) fusion convs (implicit-GEMM WMMA)
  conv3x3_wmma<256, false><<<NPIX / 16, 128, 0, stream>>>(cat, wf1p, bf1, mid, nullptr);
  conv3x3_wmma<64, true><<<NPIX / 16, 128, 0, stream>>>(mid, wf2p, bf2, nullptr, (float*)d_out);
}
